// QuantumTemporalFusionNetwork_50800873177580
// MI455X (gfx1250) — matrix operand order repaired and checked by reading, not compiled, not run on hardware
//
#include <hip/hip_runtime.h>

// ---------------------------------------------------------------------------
// QuantumTemporalFusionNetwork — CDNA5 (gfx1250) implementation.
// Dominant work: two 1-D convolutions recast as shifted GEMMs running on
// v_wmma_f32_16x16x32_bf16 (bf16 in, f32 accumulate), with double-buffered
// GLOBAL_LOAD_ASYNC_TO_LDS staging (ASYNCcnt) when the toolchain exposes it.
// ---------------------------------------------------------------------------

typedef __attribute__((ext_vector_type(16))) __bf16 v16bf;
typedef __attribute__((ext_vector_type(8)))  __bf16 v8bf;
typedef __attribute__((ext_vector_type(8)))  float  v8f;
typedef __attribute__((ext_vector_type(4)))  int    v4i;

#define BB    16
#define SS    2048
#define CIN   256
#define HH    512
#define T1    2044
#define T2    2042
#define DOUT  256
#define DQ    16
#define TWIN  32
#define THRESH 0.1f

__device__ __forceinline__ float sigm(float x) { return 1.0f / (1.0f + expf(-x)); }

// ---------------------------------------------------------------------------
// Async 16B global -> LDS copy (ASYNCcnt path) with synchronous fallback.
// Builtin signature (probe-discovered via diagnostic): (v4i AS1*, v4i AS3*, i32, i32)
// ---------------------------------------------------------------------------
#if defined(__has_builtin)
#if __has_builtin(__builtin_amdgcn_global_load_async_to_lds_b128)
#define HAVE_ASYNC_LDS 1
#endif
#endif

__device__ __forceinline__ void copy16_g2l(const __bf16* g, __bf16* l) {
#ifdef HAVE_ASYNC_LDS
  __builtin_amdgcn_global_load_async_to_lds_b128(
      (__attribute__((address_space(1))) v4i*)(__attribute__((address_space(1))) void*)(void*)g,
      (__attribute__((address_space(3))) v4i*)(__attribute__((address_space(3))) void*)(void*)l,
      0, 0);
#else
  *(v8bf*)l = *(const v8bf*)g;
#endif
}

__device__ __forceinline__ void wait_async0() {
#ifdef HAVE_ASYNC_LDS
#if defined(__has_builtin) && __has_builtin(__builtin_amdgcn_s_wait_asynccnt)
  __builtin_amdgcn_s_wait_asynccnt(0);
#else
  asm volatile("s_wait_asynccnt 0x0" ::: "memory");
#endif
#endif
}

// ---------------------------------------------------------------------------
// Elementwise f32 -> bf16 conversion (4 elems/thread)
// ---------------------------------------------------------------------------
__global__ void k_cvt_bf16(const float* __restrict__ src, __bf16* __restrict__ dst, int n) {
  int i = (blockIdx.x * blockDim.x + threadIdx.x) * 4;
  if (i + 3 < n) {
    dst[i + 0] = (__bf16)src[i + 0];
    dst[i + 1] = (__bf16)src[i + 1];
    dst[i + 2] = (__bf16)src[i + 2];
    dst[i + 3] = (__bf16)src[i + 3];
  } else {
    for (int j = i; j < n; ++j) dst[j] = (__bf16)src[j];
  }
}

// ---------------------------------------------------------------------------
// Build Keff[k][h][c] = Wr[h][c][k]*cos(ph[k]) - Wi[h][c][k]*sin(ph[k]) in bf16
// ---------------------------------------------------------------------------
__global__ void k_prep_w(const float* __restrict__ Wr, const float* __restrict__ Wi,
                         const float* __restrict__ ph, __bf16* __restrict__ Kbf,
                         int H, int C, int KW) {
  int i = blockIdx.x * blockDim.x + threadIdx.x;
  int total = KW * H * C;
  if (i >= total) return;
  int k = i / (H * C);
  int rem = i - k * H * C;
  int h = rem / C;
  int c = rem - h * C;
  float p = ph[k];
  size_t s = ((size_t)h * C + c) * KW + k;
  float v = Wr[s] * cosf(p) - Wi[s] * sinf(p);
  Kbf[i] = (__bf16)v;
}

__global__ void k_zero(float* __restrict__ p, int n) {
  int i = blockIdx.x * blockDim.x + threadIdx.x;
  if (i < n) p[i] = 0.0f;
}

// ---------------------------------------------------------------------------
// WMMA fragment loaders (ISA 7.12.2 layouts, bf16, 16x16x32)
//   A (MxK=16x32): lane = M row, K slots {kb..kb+7, 16+kb..16+kb+7}, kb=8*(lane>=16)
//   B (KxN=32x16): lane = N col, same K slots as A: {kb..kb+7, 16+kb..16+kb+7}
// Staged in LDS as [row][k], row stride LDT=40 bf16 (80B = 20 banks, conflict-free).
// ---------------------------------------------------------------------------
#define LDT 40

__device__ __forceinline__ v16bf load_a_frag(const __bf16* tile, int lane) {
  int r  = lane & 15;
  int kb = (lane >> 4) << 3;               // 0 or 8
  const __bf16* p = tile + r * LDT + kb;
  v8bf lo = *(const v8bf*)(p);             // K = kb .. kb+7
  v8bf hi = *(const v8bf*)(p + 16);        // K = 16+kb .. 16+kb+7
  v16bf f;
#pragma unroll
  for (int i = 0; i < 8; ++i) { f[i] = lo[i]; f[i + 8] = hi[i]; }
  return f;
}

__device__ __forceinline__ v16bf load_b_frag(const __bf16* tile, int lane) {
  int n  = lane & 15;
  int kb = (lane >> 4) << 3;               // 0 or 8
  const __bf16* p = tile + n * LDT + kb;
  v8bf lo = *(const v8bf*)(p);
  v8bf hi = *(const v8bf*)(p + 16);
  v16bf f;
#pragma unroll
  for (int i = 0; i < 8; ++i) { f[i] = lo[i]; f[i + 8] = hi[i]; }
  return f;
}

// ---------------------------------------------------------------------------
// Shifted-GEMM convolution:
//   Out[b, t, n] = sum_{sh<nShift} sum_{c<K} A[b, t+sh, c] * W[sh, n, c]
// Block: 256 threads (8 waves). Block tile: M=64 (time) x N=256 (channels).
// Wave tile: 32x64 = 2x4 WMMA accumulators (8 v_wmma per K-step per wave).
// K staged 32-wide through double-buffered LDS with async global->LDS copies.
// grid.x = B*32 (time tiles), grid.y = 2 (channel tiles of 256).
// ---------------------------------------------------------------------------
__global__ void __launch_bounds__(256)
k_convgemm(const __bf16* __restrict__ A, const __bf16* __restrict__ W,
           __bf16* __restrict__ Out, int Tin, int Tout, int K, int nShift) {
  __shared__ __bf16 As[2][64 * LDT];
  __shared__ __bf16 Bs[2][256 * LDT];

  const int b  = blockIdx.x >> 5;
  const int t0 = (blockIdx.x & 31) * 64;
  const int n0 = blockIdx.y * 256;
  const int tid  = threadIdx.x;
  const int wid  = tid >> 5;
  const int lane = tid & 31;
  const int waveM = wid & 1;     // 2 waves over M (32 each)
  const int waveN = wid >> 1;    // 4 waves over N (64 each)

  v8f acc[2][4];
#pragma unroll
  for (int i = 0; i < 2; ++i)
#pragma unroll
    for (int j = 0; j < 4; ++j)
#pragma unroll
      for (int e = 0; e < 8; ++e) acc[i][j][e] = 0.0f;

  const __bf16* Ab = A + (size_t)b * Tin * K;
  const int fr = tid >> 2;             // 0..63 row for LDS fills
  const int fc = (tid & 3) << 3;       // 0,8,16,24 col group
  const int kSteps = K >> 5;           // K/32
  const int nsteps = nShift * kSteps;

  // issue the async fill for one K-step into buffer `bufi`
  auto fill = [&](int step, int bufi) {
    int sh = step / kSteps;
    int c0 = (step - sh * kSteps) << 5;
    // A tile: 64 x 32
    int t = t0 + fr + sh;
    if (t > Tin - 1) t = Tin - 1;                       // clamp: feeds only masked rows
    copy16_g2l(Ab + (size_t)t * K + c0 + fc, &As[bufi][fr * LDT + fc]);
    // B tile: 256 x 32 (weights, row = output channel)
    const __bf16* Wsh = W + (size_t)sh * HH * K;
#pragma unroll
    for (int q = 0; q < 4; ++q) {
      int r = fr + q * 64;
      copy16_g2l(Wsh + (size_t)(n0 + r) * K + c0 + fc, &Bs[bufi][r * LDT + fc]);
    }
  };

  fill(0, 0);
  for (int s = 0; s < nsteps; ++s) {
    const int cur = s & 1;
    wait_async0();        // own fills of buf[cur] landed in LDS
    __syncthreads();      // everyone's fills visible; everyone done reading buf[cur^1]
    if (s + 1 < nsteps) fill(s + 1, cur ^ 1);   // overlaps with compute below

    v16bf af[2], bfr[4];
#pragma unroll
    for (int i = 0; i < 2; ++i)
      af[i] = load_a_frag(&As[cur][(waveM * 32 + i * 16) * LDT], lane);
#pragma unroll
    for (int j = 0; j < 4; ++j)
      bfr[j] = load_b_frag(&Bs[cur][(waveN * 64 + j * 16) * LDT], lane);

#pragma unroll
    for (int i = 0; i < 2; ++i)
#pragma unroll
      for (int j = 0; j < 4; ++j)
        acc[i][j] = __builtin_amdgcn_wmma_f32_16x16x32_bf16(
            false, af[i], false, bfr[j], (short)0, acc[i][j], false, false);
  }

  // ---- store C tiles as bf16 [b][t][n]: VGPR r -> M = r + 8*(lane>=16), N = lane&15
  const int rowsel = lane >> 4;
  const int col    = lane & 15;
#pragma unroll
  for (int i = 0; i < 2; ++i) {
#pragma unroll
    for (int j = 0; j < 4; ++j) {
      int hcol = n0 + waveN * 64 + j * 16 + col;
#pragma unroll
      for (int r = 0; r < 8; ++r) {
        int t = t0 + waveM * 32 + i * 16 + r + 8 * rowsel;
        if (t < Tout)
          Out[((size_t)b * Tout + t) * HH + hcol] = (__bf16)acc[i][j][r];
      }
    }
  }
}

// ---------------------------------------------------------------------------
// Per-channel precompute (H=512 blocks, 1 wave each):
//   Un = U/||U||_row ; er/ei = QS·Un ; vr/vi = e·E ; Svr/Svi/m ; cf ;
//   phase tables are32/aim32[h][w] for w = t % 32.
// ---------------------------------------------------------------------------
__global__ void k_prep_h(const float* __restrict__ QSr, const float* __restrict__ QSi,
                         const float* __restrict__ U,   const float* __restrict__ PA,
                         const float* __restrict__ E,   const float* __restrict__ CG,
                         float* __restrict__ m_out, float* __restrict__ cf_out,
                         float* __restrict__ are32, float* __restrict__ aim32) {
  const int h = blockIdx.x;
  const int lane = threadIdx.x;
  __shared__ float Un[DQ][DQ];
  __shared__ float er[DQ], ei[DQ], vr[DQ], vi[DQ];
  __shared__ float Svr_s, Svi_s;

  const float* Uh = U + (size_t)h * DQ * DQ;
  if (lane < DQ) {
    float nrm = 0.0f;
    for (int e = 0; e < DQ; ++e) { float u = Uh[lane * DQ + e]; nrm += u * u; }
    float inv = 1.0f / sqrtf(nrm);
    for (int e = 0; e < DQ; ++e) Un[lane][e] = Uh[lane * DQ + e] * inv;
  }
  __syncthreads();
  if (lane < DQ) {
    float sr = 0.0f, si = 0.0f;
    for (int d = 0; d < DQ; ++d) {
      sr += QSr[h * DQ + d] * Un[d][lane];
      si += QSi[h * DQ + d] * Un[d][lane];
    }
    er[lane] = sr; ei[lane] = si;
  }
  __syncthreads();
  if (lane < DQ) {
    const float* Eh = E + (size_t)h * DQ * DQ;
    float r = 0.0f, ii = 0.0f;
    for (int e = 0; e < DQ; ++e) {
      r  += er[e] * Eh[e * DQ + lane];
      ii += ei[e] * Eh[e * DQ + lane];
    }
    vr[lane] = r; vi[lane] = ii;
  }
  __syncthreads();
  if (lane == 0) {
    float svr = 0.0f, svi = 0.0f, mm = 0.0f;
    for (int f = 0; f < DQ; ++f) {
      svr += vr[f]; svi += vi[f];
      mm += sqrtf(vr[f] * vr[f] + vi[f] * vi[f]);
    }
    Svr_s = svr; Svi_s = svi;
    m_out[h] = mm;
    float c = 0.0f;
    for (int d = 0; d < DQ; ++d) c += sigm(CG[h * DQ + d]);
    cf_out[h] = c * (1.0f / (float)DQ);
  }
  __syncthreads();
  {
    float phi = PA[h * TWIN + lane];
    float cc = cosf(phi), ss = sinf(phi);
    are32[h * TWIN + lane] = cc * Svr_s - ss * Svi_s;
    aim32[h * TWIN + lane] = ss * Svr_s + cc * Svi_s;
  }
}

// ---------------------------------------------------------------------------
// qadv = mean_h trace(|E_h|)  (single block reduction)
// ---------------------------------------------------------------------------
__global__ void k_qadv(const float* __restrict__ E, float* __restrict__ out) {
  __shared__ float red[256];
  float s = 0.0f;
  for (int i = threadIdx.x; i < HH * DQ; i += 256) {
    int h = i >> 4, d = i & 15;
    s += fabsf(E[((size_t)h * DQ + d) * DQ + d]);
  }
  red[threadIdx.x] = s;
  __syncthreads();
  for (int o = 128; o > 0; o >>= 1) {
    if (threadIdx.x < o) red[threadIdx.x] += red[threadIdx.x + o];
    __syncthreads();
  }
  if (threadIdx.x == 0) out[0] = red[0] / (float)HH;
}

// ---------------------------------------------------------------------------
// Fused epilogue: scale = 1+0.1*tanh(mean_h h2); combined -> seq_out (d_out),
// coherence_levels, and per-b sum over t of combined into accS (for final GEMM).
// grid.x = B*32, each block handles 64 t values with 8 waves.
// ---------------------------------------------------------------------------
__global__ void __launch_bounds__(256)
k_epilogue(const __bf16* __restrict__ h2, const float* __restrict__ m,
           const float* __restrict__ cf, const float* __restrict__ are32,
           const float* __restrict__ aim32, float* __restrict__ accS,
           float* __restrict__ dout) {
  const int b  = blockIdx.x >> 5;
  const int t0 = (blockIdx.x & 31) * 64;
  const int wid = threadIdx.x >> 5, lane = threadIdx.x & 31;
  __shared__ float red[HH];
  red[threadIdx.x] = 0.0f;
  red[threadIdx.x + 256] = 0.0f;
  __syncthreads();

  float acc[16];
#pragma unroll
  for (int i = 0; i < 16; ++i) acc[i] = 0.0f;

  float* seq = dout + BB * DOUT;                               // seq_out region
  float* clv = dout + BB * DOUT + (size_t)BB * T2 * HH;        // coherence_levels

  for (int it = 0; it < 8; ++it) {
    int t = t0 + wid + it * 8;
    if (t >= T2) continue;
    // mean over H of h2 row (bf16)
    const __bf16* row = h2 + ((size_t)b * T2 + t) * HH + lane * 16;
    v8bf r0 = *(const v8bf*)(row);
    v8bf r1 = *(const v8bf*)(row + 8);
    float s = 0.0f;
#pragma unroll
    for (int i = 0; i < 8; ++i) s += (float)r0[i] + (float)r1[i];
    for (int o = 16; o > 0; o >>= 1) s += __shfl_xor(s, o, 32);
    float scale = 1.0f + 0.1f * tanhf(s * (1.0f / (float)HH));

    int w = t & (TWIN - 1);
    float* orow = seq + ((size_t)b * T2 + t) * HH;
#pragma unroll
    for (int i = 0; i < 16; ++i) {
      int h = lane + 32 * i;
      float sr  = scale * are32[h * TWIN + w];
      float si  = scale * aim32[h * TWIN + w];
      float coh = scale * m[h];
      float comb = (sigm(sr) + sigm(-sr) + sigm(si) +
                    (coh > THRESH ? 1.0f : 0.0f)) * 0.25f * cf[h];
      orow[h] = comb;
      acc[i] += comb;
    }
    if (lane == 0) {
      // ||combined/||combined|| || == 1 up to rounding; ev = 1 + 1e-12
      float ev = 1.0f + 1e-12f;
      clv[(size_t)t * BB + b] = -(ev * logf(ev));
    }
  }

  // reduce per-thread partials across the 8 waves (same h = lane + 32*i)
#pragma unroll
  for (int i = 0; i < 16; ++i) atomicAdd(&red[lane + 32 * i], acc[i]);
  __syncthreads();
  atomicAdd(&accS[b * HH + threadIdx.x], red[threadIdx.x]);
  atomicAdd(&accS[b * HH + threadIdx.x + 256], red[threadIdx.x + 256]);
}

// ---------------------------------------------------------------------------
// out[b, o] = (accS[b,:]/T2) . Wout[o,:] + bout[o]   (tiny GEMM)
// ---------------------------------------------------------------------------
__global__ void k_out(const float* __restrict__ accS, const float* __restrict__ Wout,
                      const float* __restrict__ bout, float* __restrict__ out) {
  int b = blockIdx.x, o = threadIdx.x;
  const float* a = accS + b * HH;
  const float* wrow = Wout + (size_t)o * HH;
  float s = 0.0f;
  for (int h = 0; h < HH; ++h) s += a[h] * wrow[h];
  out[b * DOUT + o] = s * (1.0f / (float)T2) + bout[o];
}

// ---------------------------------------------------------------------------
// Launch
// ---------------------------------------------------------------------------
extern "C" void kernel_launch(void* const* d_in, const int* in_sizes, int n_in,
                              void* d_out, int out_size, void* d_ws, size_t ws_size,
                              hipStream_t stream) {
  const float* x    = (const float*)d_in[0];
  const float* Wr1  = (const float*)d_in[1];
  const float* Wi1  = (const float*)d_in[2];
  const float* ph1  = (const float*)d_in[3];
  const float* Wr2  = (const float*)d_in[4];
  const float* Wi2  = (const float*)d_in[5];
  const float* ph2  = (const float*)d_in[6];
  const float* QSr  = (const float*)d_in[7];
  const float* QSi  = (const float*)d_in[8];
  const float* U    = (const float*)d_in[9];
  const float* PA   = (const float*)d_in[10];
  const float* E    = (const float*)d_in[11];
  const float* CG   = (const float*)d_in[12];
  const float* Wout = (const float*)d_in[13];
  const float* bout = (const float*)d_in[14];
  float* out = (float*)d_out;

  // ---- workspace layout (bytes) ----
  char* ws = (char*)d_ws;
  size_t o = 0;
  auto alloc = [&](size_t bytes) { size_t r = o; o += (bytes + 255) & ~(size_t)255; return r; };
  __bf16* xbf  = (__bf16*)(ws + alloc((size_t)BB * SS * CIN * 2));
  __bf16* k1bf = (__bf16*)(ws + alloc((size_t)5 * HH * CIN * 2));
  __bf16* k2bf = (__bf16*)(ws + alloc((size_t)3 * HH * HH * 2));
  __bf16* h1   = (__bf16*)(ws + alloc((size_t)BB * T1 * HH * 2));
  __bf16* h2   = (__bf16*)(ws + alloc((size_t)BB * T2 * HH * 2));
  float* m_ws  = (float*)(ws + alloc(HH * 4));
  float* cf_ws = (float*)(ws + alloc(HH * 4));
  float* are32 = (float*)(ws + alloc(HH * TWIN * 4));
  float* aim32 = (float*)(ws + alloc(HH * TWIN * 4));
  float* accS  = (float*)(ws + alloc(BB * HH * 4));

  // 1) convert x to bf16
  {
    int n = BB * SS * CIN;
    k_cvt_bf16<<<(n / 4 + 255) / 256, 256, 0, stream>>>(x, xbf, n);
  }
  // 2) effective conv kernels in bf16
  {
    int n1 = 5 * HH * CIN;
    k_prep_w<<<(n1 + 255) / 256, 256, 0, stream>>>(Wr1, Wi1, ph1, k1bf, HH, CIN, 5);
    int n2 = 3 * HH * HH;
    k_prep_w<<<(n2 + 255) / 256, 256, 0, stream>>>(Wr2, Wi2, ph2, k2bf, HH, HH, 3);
  }
  // 3) per-channel precompute + qadv
  k_prep_h<<<HH, 32, 0, stream>>>(QSr, QSi, U, PA, E, CG, m_ws, cf_ws, are32, aim32);
  k_qadv<<<1, 256, 0, stream>>>(E, out + BB * DOUT + (size_t)BB * T2 * HH + (size_t)T2 * BB);
  // 4) zero the per-b accumulator
  k_zero<<<(BB * HH + 255) / 256, 256, 0, stream>>>(accS, BB * HH);
  // 5) conv1: (B,S,256) * Keff1 -> h1 (B,2044,512), 5 shifts
  k_convgemm<<<dim3(BB * 32, 2), 256, 0, stream>>>(xbf, k1bf, h1, SS, T1, CIN, 5);
  // 6) conv2: h1 * Keff2 -> h2 (B,2042,512), 3 shifts
  k_convgemm<<<dim3(BB * 32, 2), 256, 0, stream>>>(h1, k2bf, h2, T1, T2, HH, 3);
  // 7) fused epilogue: seq_out + coherence_levels + accS
  k_epilogue<<<BB * 32, 256, 0, stream>>>(h2, m_ws, cf_ws, are32, aim32, accS, out);
  // 8) final projection
  k_out<<<BB, DOUT, 0, stream>>>(accS, Wout, bout, out);
}
